// WindowAttention_49014166782280
// MI455X (gfx1250) — matrix operand order repaired and checked by reading, not compiled, not run on hardware
//
#include <hip/hip_runtime.h>

typedef __attribute__((ext_vector_type(16))) _Float16 v16h;
typedef __attribute__((ext_vector_type(8)))  _Float16 v8h;
typedef __attribute__((ext_vector_type(8)))  float    v8f;

#define DIM   384
#define NH    12
#define HD    32
#define NSEQ  49
#define NWIN  1024
#define SCALE 0.17677669529663687f   // 32^-0.5

// padded LDS strides (elements)
#define XSTR 392   // f16 stride for 384-wide rows (x, q, k, out)
#define VSTR 72    // f16 stride for v^T rows (64 seq cols padded)
#define ASTR 68    // f32 stride for score rows
#define PSTR 72    // f16 stride for prob rows

// dynamic LDS layout (bytes)
#define OFF_X   0         // 64*392*2 = 50176   (aliased as f16 out buffer)
#define OFF_Q   50176     // 64*392*2
#define OFF_K   100352    // 64*392*2
#define OFF_VT  150528    // 384*72*2 = 55296
#define OFF_SC  205824    // 64*68*4  = 17408
#define OFF_PH  223232    // 64*72*2  = 9216
#define SMEM_BYTES 232448

__device__ __forceinline__ v16h ldA(const _Float16* p) {  // A frag: lo at +0, hi at +16
  v16h r;
  *(v8h*)&r       = *(const v8h*)p;
  *((v8h*)&r + 1) = *(const v8h*)(p + 16);
  return r;
}
__device__ __forceinline__ v16h ldB(const _Float16* p) {  // B frag: lo at +0, hi at +16
  v16h r;
  *(v8h*)&r       = *(const v8h*)p;
  *((v8h*)&r + 1) = *(const v8h*)(p + 16);
  return r;
}
#define WMMA(a, b, c) __builtin_amdgcn_wmma_f32_16x16x32_f16(false, (a), false, (b), (short)0, (c), false, false)

// ---- prologue: transpose + f32->f16 weights:  wT[c][k] = w[k][c] ----
__global__ void prep_wT(const float* __restrict__ w, _Float16* __restrict__ wT,
                        int K, int Ncol) {
  int i = blockIdx.x * 256 + threadIdx.x;
  if (i >= K * Ncol) return;
  int c = i / K;
  int k = i - c * K;
  wT[(size_t)c * K + k] = (_Float16)w[(size_t)k * Ncol + c];
}

// ---- prologue: gather relative-position bias: rpb[h][i][j] ----
__global__ void prep_rpb(const float* __restrict__ bias_table,
                         const int* __restrict__ rel_idx, float* __restrict__ rpb) {
  int i = blockIdx.x * 256 + threadIdx.x;
  if (i >= NH * NSEQ * NSEQ) return;
  int h = i / (NSEQ * NSEQ);
  int r = i - h * (NSEQ * NSEQ);
  rpb[i] = bias_table[rel_idx[r] * NH + h];
}

// ---- fused window attention: one window per workgroup, 8 waves ----
__global__ __launch_bounds__(256, 1)
void swin_attn_kernel(const float* __restrict__ x,
                      const float* __restrict__ mask,
                      const float* __restrict__ qkv_b,
                      const float* __restrict__ proj_b,
                      const _Float16* __restrict__ wT,   // [1152][384] f16 (qkv_w^T)
                      const _Float16* __restrict__ pT,   // [384][384]  f16 (proj_w^T)
                      const float* __restrict__ rpb,     // [12][49][49]
                      float* __restrict__ out) {
  extern __shared__ char smem[];
  _Float16* xh   = (_Float16*)(smem + OFF_X);
  _Float16* outh = xh;                         // reuse x buffer after QKV gemm
  _Float16* qs   = (_Float16*)(smem + OFF_Q);
  _Float16* ks   = (_Float16*)(smem + OFF_K);
  _Float16* vT   = (_Float16*)(smem + OFF_VT);
  float*    sc   = (float*)   (smem + OFF_SC);
  _Float16* ph   = (_Float16*)(smem + OFF_PH);

  const int b    = blockIdx.x;
  const int tid  = threadIdx.x;
  const int wave = tid >> 5;
  const int lane = tid & 31;
  const int l15  = lane & 15;
  const int hi   = lane >> 4;      // 0 | 1
  const int koffA  = hi * 8;       // A-fragment K offset
  const int koffB  = hi * 8;       // B-fragment K offset
  const int rowAdd = hi * 8;       // D-tile row offset

  // zero padding rows of x (rows 49..63) and whole prob buffer (once)
  for (int i = tid; i < 15 * XSTR; i += 256) xh[NSEQ * XSTR + i] = (_Float16)0.f;
  for (int i = tid; i < 64 * PSTR; i += 256) ph[i] = (_Float16)0.f;

  // load x[b] (49x384 f32) -> f16 LDS
  const float4* xg = (const float4*)(x + (size_t)b * NSEQ * DIM);
  for (int i = tid; i < NSEQ * DIM / 4; i += 256) {
    float4 f = xg[i];
    int e = i * 4;
    int n = e / DIM, c = e - n * DIM;
    _Float16* d = xh + n * XSTR + c;
    d[0] = (_Float16)f.x; d[1] = (_Float16)f.y;
    d[2] = (_Float16)f.z; d[3] = (_Float16)f.w;
  }
  __syncthreads();

  // ---- QKV GEMM: (64x384) x (384x1152); 72 col tiles, 9 per wave ----
  for (int ctl = 0; ctl < 9; ++ctl) {
    const int cb  = (wave * 9 + ctl) * 16;
    const int col = cb + l15;
    const _Float16* wcol = wT + (size_t)col * DIM;
    v8f acc[4] = {};
    for (int kc = 0; kc < 12; ++kc) {
      const int kb = kc * 32;
      v16h bf = ldB(wcol + kb + koffB);
      const _Float16* abase = xh + l15 * XSTR + kb + koffA;
#pragma unroll
      for (int rt = 0; rt < 4; ++rt) {
        v16h af = ldA(abase + rt * 16 * XSTR);
        acc[rt] = WMMA(af, bf, acc[rt]);
      }
    }
    const float bias = qkv_b[col];
    if (cb < DIM) {                       // Q: row-major f16, 1/sqrt(d) folded in
      _Float16* d0 = qs + cb + l15;
#pragma unroll
      for (int rt = 0; rt < 4; ++rt)
#pragma unroll
        for (int i = 0; i < 8; ++i)
          d0[(rt * 16 + i + rowAdd) * XSTR] = (_Float16)((acc[rt][i] + bias) * SCALE);
    } else if (cb < 2 * DIM) {            // K: row-major f16
      _Float16* d0 = ks + (cb - DIM) + l15;
#pragma unroll
      for (int rt = 0; rt < 4; ++rt)
#pragma unroll
        for (int i = 0; i < 8; ++i)
          d0[(rt * 16 + i + rowAdd) * XSTR] = (_Float16)(acc[rt][i] + bias);
    } else {                              // V: transposed -> vT[d][n]
      _Float16* d0 = vT + (cb - 2 * DIM + l15) * VSTR;
#pragma unroll
      for (int rt = 0; rt < 4; ++rt)
#pragma unroll
        for (int i = 0; i < 8; ++i)
          d0[rt * 16 + i + rowAdd] = (_Float16)(acc[rt][i] + bias);
    }
  }
  __syncthreads();

  // softmax work distribution: 4 lanes per row, all 256 threads
  const int srow_id = tid >> 2;   // 0..63
  const int ssub    = tid & 3;    // 0..3

  // ---- attention, one head at a time ----
  for (int h = 0; h < NH; ++h) {
    const int hb = h * HD;
    // (a) S = (Q*s) K^T : 16 tiles, 2 per wave (K = HD = 32, single WMMA)
#pragma unroll
    for (int s = 0; s < 2; ++s) {
      const int t  = wave * 2 + s;
      const int rt = t >> 2, cj = t & 3;
      v16h af = ldA(qs + (rt * 16 + l15) * XSTR + hb + koffA);
      v16h bf = ldB(ks + (cj * 16 + l15) * XSTR + hb + koffB);
      v8f c = {};
      c = WMMA(af, bf, c);
      float* so = sc + (rt * 16 + rowAdd) * ASTR + cj * 16 + l15;
#pragma unroll
      for (int i = 0; i < 8; ++i) so[i * ASTR] = c[i];
    }
    __syncthreads();
    // (b) softmax: +rel-pos bias +window mask; 4 lanes cooperate per row
    if (srow_id < NSEQ) {
      const float* mrow = mask + (size_t)(b & (NWIN - 1)) * NSEQ * NSEQ + srow_id * NSEQ;
      const float* rrow = rpb + (size_t)h * NSEQ * NSEQ + srow_id * NSEQ;
      float* srow = sc + srow_id * ASTR;
      float mx = -1e30f;
      for (int j = ssub; j < NSEQ; j += 4) {
        float v = srow[j] + rrow[j] + mrow[j];
        srow[j] = v;
        mx = fmaxf(mx, v);
      }
      mx = fmaxf(mx, __shfl_xor(mx, 1, 32));
      mx = fmaxf(mx, __shfl_xor(mx, 2, 32));
      float sum = 0.f;
      for (int j = ssub; j < NSEQ; j += 4) {
        float e = __expf(srow[j] - mx);
        srow[j] = e;
        sum += e;
      }
      sum += __shfl_xor(sum, 1, 32);
      sum += __shfl_xor(sum, 2, 32);
      const float inv = 1.f / sum;
      _Float16* prow = ph + srow_id * PSTR;
      for (int j = ssub; j < NSEQ; j += 4) prow[j] = (_Float16)(srow[j] * inv);
    }
    __syncthreads();
    // (c) O = P V : 4 row-tiles x 2 col-tiles = 8 tiles, 1 per wave (K = 64)
    {
      const int rt = wave >> 1, cd = wave & 1;
      v8f c = {};
#pragma unroll
      for (int kc = 0; kc < 2; ++kc) {
        v16h af = ldA(ph + (rt * 16 + l15) * PSTR + kc * 32 + koffA);
        v16h bf = ldB(vT + (hb + cd * 16 + l15) * VSTR + kc * 32 + koffB);
        c = WMMA(af, bf, c);
      }
      _Float16* oo = outh + hb + cd * 16 + l15;
#pragma unroll
      for (int i = 0; i < 8; ++i)
        oo[(rt * 16 + i + rowAdd) * XSTR] = (_Float16)c[i];
    }
  }
  __syncthreads();

  // ---- projection GEMM: (64x384) x (384x384), 24 col tiles, 3 per wave ----
  for (int ctl = 0; ctl < 3; ++ctl) {
    const int cb  = (wave * 3 + ctl) * 16;
    const int col = cb + l15;
    const _Float16* pcol = pT + (size_t)col * DIM;
    v8f acc[4] = {};
    for (int kc = 0; kc < 12; ++kc) {
      const int kb = kc * 32;
      v16h bf = ldB(pcol + kb + koffB);
      const _Float16* abase = outh + l15 * XSTR + kb + koffA;
#pragma unroll
      for (int rt = 0; rt < 4; ++rt) {
        v16h af = ldA(abase + rt * 16 * XSTR);
        acc[rt] = WMMA(af, bf, acc[rt]);
      }
    }
    const float bias = proj_b[col];
    float* og = out + (size_t)b * NSEQ * DIM + col;
#pragma unroll
    for (int rt = 0; rt < 4; ++rt)
#pragma unroll
      for (int i = 0; i < 8; ++i) {
        const int r = rt * 16 + i + rowAdd;
        if (r < NSEQ) og[r * DIM] = acc[rt][i] + bias;
      }
  }
}

extern "C" void kernel_launch(void* const* d_in, const int* in_sizes, int n_in,
                              void* d_out, int out_size, void* d_ws, size_t ws_size,
                              hipStream_t stream) {
  (void)in_sizes; (void)n_in; (void)out_size; (void)ws_size;
  const float* x          = (const float*)d_in[0];
  const float* mask       = (const float*)d_in[1];
  const float* qkv_w      = (const float*)d_in[2];
  const float* qkv_b      = (const float*)d_in[3];
  const float* bias_table = (const float*)d_in[4];
  const float* proj_w     = (const float*)d_in[5];
  const float* proj_b     = (const float*)d_in[6];
  const int*   rel_idx    = (const int*)d_in[7];
  float* out = (float*)d_out;

  char* ws = (char*)d_ws;
  _Float16* wT  = (_Float16*)ws;                                   // 1152*384 f16
  _Float16* pT  = (_Float16*)(ws + 1152 * 384 * 2);                // 384*384 f16
  float*    rpb = (float*)(ws + 1152 * 384 * 2 + 384 * 384 * 2);   // 12*49*49 f32

  prep_wT<<<(1152 * 384 + 255) / 256, 256, 0, stream>>>(qkv_w, wT, DIM, 3 * DIM);
  prep_wT<<<(384 * 384 + 255) / 256, 256, 0, stream>>>(proj_w, pT, DIM, DIM);
  prep_rpb<<<(NH * NSEQ * NSEQ + 255) / 256, 256, 0, stream>>>(bias_table, rel_idx, rpb);

  hipFuncSetAttribute((const void*)swin_attn_kernel,
                      hipFuncAttributeMaxDynamicSharedMemorySize, SMEM_BYTES);
  swin_attn_kernel<<<4096, 256, SMEM_BYTES, stream>>>(
      x, mask, qkv_b, proj_b, wT, pT, rpb, out);
}
